// PolymerJEPA_90486370992336
// MI455X (gfx1250) — compile-verified, matrix operand order checked by reading, not run on hardware
//
#include <hip/hip_runtime.h>
#include <cstdint>
#include <cstddef>

typedef __attribute__((ext_vector_type(16))) _Float16 v16h;
typedef __attribute__((ext_vector_type(8)))  float    v8f;

#define CDIV(a,b) (((a)+(b)-1)/(b))

// ---------------- constants ----------------
static constexpr int Bn = 256, Pn = 32, NTPc = 4, HID = 128;
static constexpr int N_NODESc = 80000, N_SUBc = 200000, E_SUBc = 600000;
static constexpr int NF_EDGEc = 16;
static constexpr int BP = Bn * Pn; // 8192

// output offsets (floats)
static constexpr size_t OFF_ETGT  = 0;
static constexpr size_t OFF_PRED  = 131072;
static constexpr size_t OFF_VCTX1 = 262144;
static constexpr size_t OFF_VINIT = 294912;
static constexpr size_t OFF_VCTX2 = 327680;
static constexpr size_t OFF_VTGT  = 360448;
static constexpr size_t OFF_VGRPH = 393216;

// ---------------- WMMA GEMM ----------------
// out[M x N] = epilogue( Arows @ W[KW x N] )  with f16 operands, f32 accum.
// Block tile: 64 (M) x 128 (N); 8 wave32 waves, each wave owns one 16-col
// strip and 4 M-subtiles -> 4 v_wmma_f32_16x16x32_f16 per K-step.
// Double-buffered LDS (ping-pong As/Bt): one barrier per K-step, stage of
// tile k+1 overlaps WMMA of tile k. Gather row pointers hoisted out of the
// K loop (a_idx/a2_idx read once per thread).
// A row r: cols [0,K1) from A[(HAS_IDX?a_idx[r]:r)*K1], cols [K1,K1+K2) from
// A2[a2_idx[r]*K2] (fused gather/concat), zero-padded to mult of 32.
// All K1/K2 are multiples of 4 -> staging is pure float4 loads.
// flags bit0: relu before addend; bit1: relu after addend.
// rscale[rs_idx[r]] row scale; dst_idx -> atomicAdd scatter into out.
template <bool HAS_IDX, bool HAS_A2>
__global__ __launch_bounds__(256) void wmma_gemm_kernel(
    const float* __restrict__ A, int K1,
    const int* __restrict__ a_idx,
    const float* __restrict__ A2, const int* __restrict__ a2_idx, int K2,
    const _Float16* __restrict__ W, int KW, int N,
    const float* __restrict__ bias,
    const float* __restrict__ addend,
    const float* __restrict__ rscale, const int* __restrict__ rs_idx,
    const int* __restrict__ dst_idx,
    float* __restrict__ out, int M, int flags)
{
  __shared__ __align__(32) _Float16 As[2][64][32];
  __shared__ __align__(32) _Float16 Bt[2][128][32]; // transposed weight tiles
  const int tid  = threadIdx.x;
  const int wid  = tid >> 5;
  const int lane = tid & 31;
  const int row0 = blockIdx.x * 64;
  const int colBase = blockIdx.y * 128;
  const int col0 = colBase + wid * 16;
  v8f acc0 = {}, acc1 = {}, acc2 = {}, acc3 = {};

  // ---- hoist per-slot A row pointers (K-loop invariant)
  const float* aRow[2];
  const float* a2Row[2];
  int  c4s[2], rIdx[2];
  bool rowOK[2];
  #pragma unroll
  for (int j = 0; j < 2; ++j) {
    int slot = tid + j * 256;          // 512 float4-slots
    int r  = slot >> 3;                // 0..63
    int c4 = (slot & 7) * 4;           // 0..28
    int gr = row0 + r;
    rIdx[j] = r; c4s[j] = c4;
    rowOK[j] = gr < M;
    int grc = rowOK[j] ? gr : 0;
    int ar  = HAS_IDX ? a_idx[grc] : grc;
    aRow[j] = A + (size_t)ar * K1;
    a2Row[j] = HAS_A2 ? (A2 + (size_t)a2_idx[grc] * K2) : nullptr;
  }

  auto stageA = [&](int k0, int nb) {
    #pragma unroll
    for (int j = 0; j < 2; ++j) {
      int r = rIdx[j], c4 = c4s[j];
      float4 f = make_float4(0.f, 0.f, 0.f, 0.f);
      int kk = k0 + c4;
      if (rowOK[j]) {
        if (kk < K1) f = *(const float4*)(aRow[j] + kk);
        else if (HAS_A2) {
          int k2 = kk - K1;
          if (k2 < K2) f = *(const float4*)(a2Row[j] + k2);
        }
      }
      As[nb][r][c4 + 0] = (_Float16)f.x;
      As[nb][r][c4 + 1] = (_Float16)f.y;
      As[nb][r][c4 + 2] = (_Float16)f.z;
      As[nb][r][c4 + 3] = (_Float16)f.w;
    }
  };
  auto stageB = [&](int k0, int nb) {
    #pragma unroll
    for (int j = 0; j < 8; ++j) {
      int e2 = tid + j * 256;          // 2048 half-pairs
      int kk = e2 >> 6;
      int c  = (e2 & 63) * 2;
      int gk = k0 + kk;
      int gc = colBase + c;
      _Float16 w0 = (_Float16)0.f, w1 = (_Float16)0.f;
      if (gk < KW && gc < N) {
        union { unsigned int u; _Float16 h[2]; } cv;
        cv.u = *(const unsigned int*)(W + (size_t)gk * N + gc);
        w0 = cv.h[0]; w1 = cv.h[1];
      }
      Bt[nb][c][kk]     = w0;
      Bt[nb][c + 1][kk] = w1;
    }
  };

  // ---- software pipeline: stage tile 0, then {stage k+1 | compute k}
  stageA(0, 0);
  stageB(0, 0);
  __syncthreads();
  int nb = 0;
  for (int k0 = 0; k0 < KW; k0 += 32) {
    int kn = k0 + 32;
    if (kn < KW) {
      stageA(kn, nb ^ 1);
      stageB(kn, nb ^ 1);
      if (kn + 32 < KW)
        __builtin_prefetch((const void*)(W + (size_t)(kn + 32) * N + colBase), 0, 1);
    }
    {
      int rr = lane & 15;
      int kh = (lane >> 4) * 16;
      v16h b  = *(const v16h*)&Bt[nb][wid * 16 + rr][kh];
      v16h a0 = *(const v16h*)&As[nb][rr][kh];
      v16h a1 = *(const v16h*)&As[nb][16 + rr][kh];
      v16h a2 = *(const v16h*)&As[nb][32 + rr][kh];
      v16h a3 = *(const v16h*)&As[nb][48 + rr][kh];
      acc0 = __builtin_amdgcn_wmma_f32_16x16x32_f16(false, a0, false, b, (short)0, acc0, false, false);
      acc1 = __builtin_amdgcn_wmma_f32_16x16x32_f16(false, a1, false, b, (short)0, acc1, false, false);
      acc2 = __builtin_amdgcn_wmma_f32_16x16x32_f16(false, a2, false, b, (short)0, acc2, false, false);
      acc3 = __builtin_amdgcn_wmma_f32_16x16x32_f16(false, a3, false, b, (short)0, acc3, false, false);
    }
    __syncthreads();
    nb ^= 1;
  }

  // ---- epilogue (C layout: VGPR i -> M = i or i+8; N = lane&15)
  v8f accs[4] = {acc0, acc1, acc2, acc3};
  #pragma unroll
  for (int mi = 0; mi < 4; ++mi) {
    #pragma unroll
    for (int i = 0; i < 8; ++i) {
      int m = mi * 16 + ((lane < 16) ? i : (i + 8));
      int n = lane & 15;
      int gr = row0 + m;
      int gc = col0 + n;
      if (gr < M && gc < N) {
        float v = accs[mi][i] + bias[gc];
        if (flags & 1) v = fmaxf(v, 0.f);
        if (rscale) { int rridx = rs_idx ? rs_idx[gr] : gr; v *= rscale[rridx]; }
        if (addend) v += addend[(size_t)gr * N + gc];
        if (flags & 2) v = fmaxf(v, 0.f);
        if (dst_idx) atomicAdd(&out[(size_t)dst_idx[gr] * N + gc], v);
        else out[(size_t)gr * N + gc] = v;
      }
    }
  }
}

// ---------------- elementwise / segment kernels ----------------
__global__ void cvt_f16_kernel(const float* __restrict__ x, _Float16* __restrict__ y, int n) {
  int i = blockIdx.x * blockDim.x + threadIdx.x;
  if (i < n) y[i] = (_Float16)x[i];
}

__global__ void copy_kernel(const float* __restrict__ a, float* __restrict__ b, int n) {
  int i = blockIdx.x * blockDim.x + threadIdx.x;
  if (i < n) b[i] = a[i];
}

__global__ void seg_sum_kernel(const float* __restrict__ x, const int* __restrict__ idx,
                               float* __restrict__ out, int rows) {
  size_t i = (size_t)blockIdx.x * blockDim.x + threadIdx.x;
  if (i >= (size_t)rows * HID) return;
  int r = (int)(i >> 7), c = (int)(i & 127);
  atomicAdd(&out[(size_t)idx[r] * HID + c], x[i]);
}

__global__ void seg_cnt_kernel(const int* __restrict__ idx, float* __restrict__ cnt, int rows) {
  int r = blockIdx.x * blockDim.x + threadIdx.x;
  if (r < rows) atomicAdd(&cnt[idx[r]], 1.0f);
}

__global__ void seg_div_kernel(float* __restrict__ s, const float* __restrict__ cnt, int segs) {
  size_t i = (size_t)blockIdx.x * blockDim.x + threadIdx.x;
  if (i >= (size_t)segs * HID) return;
  s[i] /= fmaxf(cnt[i >> 7], 1.0f);
}

__global__ void gather_kernel(const float* __restrict__ src, const int* __restrict__ idx,
                              int idx_stride, float* __restrict__ out, int rows) {
  size_t i = (size_t)blockIdx.x * blockDim.x + threadIdx.x;
  if (i >= (size_t)rows * HID) return;
  int r = (int)(i >> 7), c = (int)(i & 127);
  out[i] = src[(size_t)idx[(size_t)r * idx_stride] * HID + c];
}

__global__ void tgt_idx_kernel(const int* __restrict__ tgt, int* __restrict__ out) {
  int i = blockIdx.x * blockDim.x + threadIdx.x;
  if (i < Bn * NTPc) out[i] = tgt[i] + (i / NTPc) * Pn;
}

__global__ void bcast_add_kernel(const float* __restrict__ base, const float* __restrict__ t,
                                 float* __restrict__ out) {
  size_t i = (size_t)blockIdx.x * blockDim.x + threadIdx.x;
  if (i >= (size_t)Bn * NTPc * HID) return;
  int r = (int)(i >> 7), c = (int)(i & 127);
  out[i] = base[(size_t)(r / NTPc) * HID + c] + t[i];
}

__global__ void ln_kernel(const float* __restrict__ x, float* __restrict__ y,
                          const float* __restrict__ g, const float* __restrict__ b,
                          int rows, int dorelu) {
  int r = blockIdx.x * blockDim.x + threadIdx.x;
  if (r >= rows) return;
  const float* xr = x + (size_t)r * HID;
  float m = 0.f;
  for (int i = 0; i < HID; ++i) m += xr[i];
  m *= (1.f / HID);
  float v = 0.f;
  for (int i = 0; i < HID; ++i) { float d = xr[i] - m; v += d * d; }
  v *= (1.f / HID);
  float inv = rsqrtf(v + 1e-5f);
  float* yr = y + (size_t)r * HID;
  for (int i = 0; i < HID; ++i) {
    float o = (xr[i] - m) * inv * g[i] + b[i];
    if (dorelu) o = fmaxf(o, 0.f);
    yr[i] = o;
  }
}

__global__ __launch_bounds__(256) void attn_kernel(
    const float* __restrict__ Q, const float* __restrict__ K, const float* __restrict__ V,
    const float* __restrict__ adj, const unsigned char* __restrict__ msk,
    float* __restrict__ out) {
  int b = blockIdx.x;
  __shared__ float s[32][32];
  const float scale = 0.08838834764831845f; // 1/sqrt(128)
  int t = threadIdx.x;
  for (int p = t; p < 32 * 32; p += 256) {
    int q = p >> 5, k = p & 31;
    const float* qp = Q + ((size_t)b * 32 + q) * HID;
    const float* kp = K + ((size_t)b * 32 + k) * HID;
    float a = 0.f;
    for (int d = 0; d < HID; ++d) a += qp[d] * kp[d];
    a = a * scale * adj[((size_t)b * 32 + q) * 32 + k];
    if (!msk[b * 32 + k]) a = -1e9f;
    s[q][k] = a;
  }
  __syncthreads();
  if (t < 32) {
    float mx = -1e30f;
    for (int k = 0; k < 32; ++k) mx = fmaxf(mx, s[t][k]);
    float sum = 0.f;
    for (int k = 0; k < 32; ++k) { float e = expf(s[t][k] - mx); s[t][k] = e; sum += e; }
    float inv = 1.0f / sum;
    for (int k = 0; k < 32; ++k) s[t][k] *= inv;
  }
  __syncthreads();
  for (int p = t; p < 32 * HID; p += 256) {
    int q = p >> 7, d = p & 127;
    float a = 0.f;
    const float* vp = V + (size_t)b * 32 * HID + d;
    for (int k = 0; k < 32; ++k) a += s[q][k] * vp[(size_t)k * HID];
    out[((size_t)b * 32 + q) * HID + d] = a;
  }
}

__global__ void masked_mean_kernel(const float* __restrict__ x, const unsigned char* __restrict__ m,
                                   float* __restrict__ out) {
  int b = blockIdx.x, d = threadIdx.x;
  float s = 0.f, c = 0.f;
  for (int p = 0; p < Pn; ++p) {
    float mm = m[b * Pn + p] ? 1.f : 0.f;
    s += x[((size_t)b * Pn + p) * HID + d] * mm;
    c += mm;
  }
  out[(size_t)b * HID + d] = s / fmaxf(c, 1e-9f);
}

// ---------------- host helpers ----------------
struct Bump {
  char* base; size_t off;
  void* take(size_t bytes) {
    size_t a = (off + 255) & ~(size_t)255;
    off = a + bytes;
    return base + a;
  }
  float*    f(size_t n) { return (float*)take(n * 4); }
  int*      i(size_t n) { return (int*)take(n * 4); }
  _Float16* h(size_t n) { return (_Float16*)take(n * 2); }
};

static void gemm(hipStream_t s, const float* A, int K1, const int* a_idx,
                 const float* A2, const int* a2_idx, int K2,
                 const _Float16* W, int KW, int N,
                 const float* bias, const float* addend,
                 const float* rscale, const int* rs_idx,
                 const int* dst_idx, float* out, int M, int flags) {
  dim3 grid(CDIV(M, 64), CDIV(N, 128));
  if (A2)
    wmma_gemm_kernel<true, true><<<grid, 256, 0, s>>>(A, K1, a_idx, A2, a2_idx, K2, W, KW, N,
                                                      bias, addend, rscale, rs_idx, dst_idx, out, M, flags);
  else if (a_idx)
    wmma_gemm_kernel<true, false><<<grid, 256, 0, s>>>(A, K1, a_idx, A2, a2_idx, K2, W, KW, N,
                                                       bias, addend, rscale, rs_idx, dst_idx, out, M, flags);
  else
    wmma_gemm_kernel<false, false><<<grid, 256, 0, s>>>(A, K1, a_idx, A2, a2_idx, K2, W, KW, N,
                                                        bias, addend, rscale, rs_idx, dst_idx, out, M, flags);
}

// param leaf indices (pytree: dicts by sorted key)
enum {
  IDX_X = 0, IDX_RW = 1, IDX_NW = 2, IDX_EA = 3, IDX_EW = 4, IDX_PE = 5,
  IDX_ADJ = 6, IDX_CMASK = 7, IDX_MASK = 8, IDX_TGT = 9, IDX_NM = 10,
  IDX_CS = 11, IDX_SEM = 12, IDX_SB = 13,
  U0W = 14, U0B = 15, U1W = 16, U1B = 17,
  CTX0 = 18, CTX1 = 34,            // 16 leaves per enc layer
  GNN0 = 50,                        // h.W h.b msg.W msg.b w0.W w0.b
  GNN1 = 56, GNN2 = 60,             // h.W h.b msg.W msg.b
  INW = 64, INB = 65,
  PL1W = 66, PL1B = 67, PL2W = 68, PL2B = 69, PLNB = 70, PLNG = 71,
  PRWW = 72, PRWB = 73, RWW = 74, RWB = 75,
  TGT0 = 76, TGT1 = 92
};
// enc layer leaf offsets: 0 f1.W 1 f1.b 2 f2.W 3 f2.b 4 k.W 5 k.b
// 6 ln1_b 7 ln1_g 8 ln2_b 9 ln2_g 10 o.W 11 o.b 12 q.W 13 q.b 14 v.W 15 v.b

extern "C" void kernel_launch(void* const* d_in, const int* in_sizes, int n_in,
                              void* d_out, int out_size, void* d_ws, size_t ws_size,
                              hipStream_t stream) {
  (void)in_sizes; (void)n_in; (void)out_size; (void)ws_size;
  auto F = [&](int i) { return (const float*)d_in[i]; };
  auto I = [&](int i) { return (const int*)d_in[i]; };
  auto Bv = [&](int i) { return (const unsigned char*)d_in[i]; };
  float* out = (float*)d_out;

  Bump ws{(char*)d_ws, 0};

  // ---- activation buffers
  float* tmp_node = ws.f((size_t)N_NODESc * HID);  // rw-MLP result; later node seg-sum
  float* h_node   = ws.f((size_t)N_NODESc * HID);
  float* xs       = ws.f((size_t)N_SUBc * HID);
  float* h0       = ws.f((size_t)N_SUBc * HID);
  float* agg      = ws.f((size_t)N_SUBc * HID);
  float* pm       = ws.f((size_t)BP * HID);
  float* cnt_pm   = ws.f(BP);
  float* cnt_node = ws.f(N_NODESc);
  float* emb      = ws.f((size_t)BP * HID);
  float* ctxb     = ws.f((size_t)BP * HID);
  float* tgtb     = ws.f((size_t)BP * HID);
  float* hln      = ws.f((size_t)BP * HID);
  float* qb       = ws.f((size_t)BP * HID);
  float* kb       = ws.f((size_t)BP * HID);
  float* vb       = ws.f((size_t)BP * HID);
  float* atn      = ws.f((size_t)BP * HID);
  float* ffn1     = ws.f((size_t)BP * 4 * HID);
  float* ectx     = ws.f((size_t)Bn * HID);
  float* tpe      = ws.f((size_t)Bn * NTPc * HID);
  float* cond     = ws.f((size_t)Bn * NTPc * HID);
  float* predh    = ws.f((size_t)Bn * NTPc * HID);
  int*   tidx     = ws.i(Bn * NTPc);

  // ---- f16 weight conversions
  auto cvt = [&](int idx, int n) -> const _Float16* {
    _Float16* p = ws.h(n);
    cvt_f16_kernel<<<CDIV(n, 256), 256, 0, stream>>>(F(idx), p, n);
    return p;
  };
  const _Float16* Win16  = cvt(INW, 64 * 128);
  const _Float16* Wrw16  = cvt(RWW, 16 * 128);
  const _Float16* Wprw16 = cvt(PRWW, 16 * 128);
  const _Float16* WU16[2]   = {cvt(U0W, 128 * 128), cvt(U1W, 128 * 128)};
  const int gnnBase[3] = {GNN0, GNN1, GNN2};
  const _Float16 *Wmsg16[3], *Wh16[3];
  for (int l = 0; l < 3; ++l) {
    Wh16[l]   = cvt(gnnBase[l] + 0, 128 * 128);
    Wmsg16[l] = cvt(gnnBase[l] + 2, 144 * 128);
  }
  const _Float16* Ww016 = cvt(GNN0 + 4, 128 * 128);
  struct EncW { const _Float16 *f1, *f2, *k, *o, *q, *v; };
  const int encBase[4] = {CTX0, CTX1, TGT0, TGT1};
  EncW enc[4];
  for (int e = 0; e < 4; ++e) {
    int bI = encBase[e];
    enc[e].f1 = cvt(bI + 0, 128 * 512);
    enc[e].f2 = cvt(bI + 2, 512 * 128);
    enc[e].k  = cvt(bI + 4, 128 * 128);
    enc[e].o  = cvt(bI + 10, 128 * 128);
    enc[e].q  = cvt(bI + 12, 128 * 128);
    enc[e].v  = cvt(bI + 14, 128 * 128);
  }
  const _Float16* Wl1 = cvt(PL1W, 128 * 128);
  const _Float16* Wl2 = cvt(PL2W, 128 * 128);

  // ---- node embedding: h = x@Win + b + relu(rw@Wrw + b)
  gemm(stream, F(IDX_RW), 16, nullptr, nullptr, nullptr, 0, Wrw16, 16, 128,
       F(RWB), nullptr, nullptr, nullptr, nullptr, tmp_node, N_NODESc, 1);
  gemm(stream, F(IDX_X), 64, nullptr, nullptr, nullptr, 0, Win16, 64, 128,
       F(INB), tmp_node, nullptr, nullptr, nullptr, h_node, N_NODESc, 0);

  // xs = h[nm]
  gather_kernel<<<CDIV((size_t)N_SUBc * HID, 256), 256, 0, stream>>>(
      h_node, I(IDX_NM), 1, xs, N_SUBc);

  const int* srcI = I(IDX_CS);
  const int* dstI = I(IDX_CS) + E_SUBc;

  // ---- GNN layers
  for (int l = 0; l < 3; ++l) {
    if (l > 0) {
      hipMemsetAsync(pm, 0, (size_t)BP * HID * 4, stream);
      hipMemsetAsync(cnt_pm, 0, BP * 4, stream);
      seg_sum_kernel<<<CDIV((size_t)N_SUBc * HID, 256), 256, 0, stream>>>(xs, I(IDX_SB), pm, N_SUBc);
      seg_cnt_kernel<<<CDIV(N_SUBc, 256), 256, 0, stream>>>(I(IDX_SB), cnt_pm, N_SUBc);
      seg_div_kernel<<<CDIV((size_t)BP * HID, 256), 256, 0, stream>>>(pm, cnt_pm, BP);
      // xs += relu(pm[sb] @ U + b)
      gemm(stream, pm, 128, I(IDX_SB), nullptr, nullptr, 0, WU16[l - 1], 128, 128,
           F(l == 1 ? U0B : U1B), xs, nullptr, nullptr, nullptr, xs, N_SUBc, 1);
      // xs = segment_mean(xs, nm, N_NODES)[nm]
      hipMemsetAsync(tmp_node, 0, (size_t)N_NODESc * HID * 4, stream);
      hipMemsetAsync(cnt_node, 0, N_NODESc * 4, stream);
      seg_sum_kernel<<<CDIV((size_t)N_SUBc * HID, 256), 256, 0, stream>>>(xs, I(IDX_NM), tmp_node, N_SUBc);
      seg_cnt_kernel<<<CDIV(N_SUBc, 256), 256, 0, stream>>>(I(IDX_NM), cnt_node, N_SUBc);
      seg_div_kernel<<<CDIV((size_t)N_NODESc * HID, 256), 256, 0, stream>>>(tmp_node, cnt_node, N_NODESc);
      gather_kernel<<<CDIV((size_t)N_SUBc * HID, 256), 256, 0, stream>>>(tmp_node, I(IDX_NM), 1, xs, N_SUBc);
    }
    // agg = scatter_add( relu([xs[src]||ea[sem]] @ Wmsg + b) * ew[sem], dst )
    hipMemsetAsync(agg, 0, (size_t)N_SUBc * HID * 4, stream);
    gemm(stream, xs, 128, srcI, F(IDX_EA), I(IDX_SEM), NF_EDGEc,
         Wmsg16[l], 144, 128, F(gnnBase[l] + 3), nullptr,
         F(IDX_EW), I(IDX_SEM), dstI, agg, E_SUBc, 1);
    if (l == 0)
      gemm(stream, xs, 128, nullptr, nullptr, nullptr, 0, Ww016, 128, 128,
           F(GNN0 + 5), nullptr, nullptr, nullptr, nullptr, h0, N_SUBc, 1);
    // xs = relu(h0 + agg @ Wh + b)
    gemm(stream, agg, 128, nullptr, nullptr, nullptr, 0, Wh16[l], 128, 128,
         F(gnnBase[l] + 1), h0, nullptr, nullptr, nullptr, xs, N_SUBc, 2);
  }

  // ---- emb = segment_mean(xs, sb, B*P)
  hipMemsetAsync(emb, 0, (size_t)BP * HID * 4, stream);
  hipMemsetAsync(cnt_pm, 0, BP * 4, stream);
  seg_sum_kernel<<<CDIV((size_t)N_SUBc * HID, 256), 256, 0, stream>>>(xs, I(IDX_SB), emb, N_SUBc);
  seg_cnt_kernel<<<CDIV(N_SUBc, 256), 256, 0, stream>>>(I(IDX_SB), cnt_pm, N_SUBc);
  seg_div_kernel<<<CDIV((size_t)BP * HID, 256), 256, 0, stream>>>(emb, cnt_pm, BP);

  tgt_idx_kernel<<<CDIV(Bn * NTPc, 256), 256, 0, stream>>>(I(IDX_TGT), tidx);

  // ---- encoders
  auto run_encoder = [&](int e0, float* xbuf, const unsigned char* msk) {
    for (int L = 0; L < 2; ++L) {
      int bI = encBase[e0 + L];
      const EncW& w = enc[e0 + L];
      ln_kernel<<<CDIV(BP, 256), 256, 0, stream>>>(xbuf, hln, F(bI + 7), F(bI + 6), BP, 0);
      gemm(stream, hln, 128, nullptr, nullptr, nullptr, 0, w.q, 128, 128, F(bI + 13),
           nullptr, nullptr, nullptr, nullptr, qb, BP, 0);
      gemm(stream, hln, 128, nullptr, nullptr, nullptr, 0, w.k, 128, 128, F(bI + 5),
           nullptr, nullptr, nullptr, nullptr, kb, BP, 0);
      gemm(stream, hln, 128, nullptr, nullptr, nullptr, 0, w.v, 128, 128, F(bI + 15),
           nullptr, nullptr, nullptr, nullptr, vb, BP, 0);
      attn_kernel<<<Bn, 256, 0, stream>>>(qb, kb, vb, F(IDX_ADJ), msk, atn);
      gemm(stream, atn, 128, nullptr, nullptr, nullptr, 0, w.o, 128, 128, F(bI + 11),
           xbuf, nullptr, nullptr, nullptr, xbuf, BP, 0);
      ln_kernel<<<CDIV(BP, 256), 256, 0, stream>>>(xbuf, hln, F(bI + 9), F(bI + 8), BP, 0);
      gemm(stream, hln, 128, nullptr, nullptr, nullptr, 0, w.f1, 128, 512, F(bI + 1),
           nullptr, nullptr, nullptr, nullptr, ffn1, BP, 1);
      gemm(stream, ffn1, 512, nullptr, nullptr, nullptr, 0, w.f2, 512, 128, F(bI + 3),
           xbuf, nullptr, nullptr, nullptr, xbuf, BP, 0);
    }
  };

  // ctx = emb + relu(pe @ Wprw + b); then ctx encoder; then masked mean
  gemm(stream, F(IDX_PE), 16, nullptr, nullptr, nullptr, 0, Wprw16, 16, 128,
       F(PRWB), emb, nullptr, nullptr, nullptr, ctxb, BP, 1);
  run_encoder(0, ctxb, Bv(IDX_CMASK));
  masked_mean_kernel<<<Bn, 128, 0, stream>>>(ctxb, Bv(IDX_CMASK), ectx);

  // tgt mixer over emb
  hipMemcpyAsync(tgtb, emb, (size_t)BP * HID * 4, hipMemcpyDeviceToDevice, stream);
  run_encoder(2, tgtb, Bv(IDX_MASK));

  // ---- outputs
  gather_kernel<<<CDIV((size_t)Bn * NTPc * HID, 256), 256, 0, stream>>>(
      tgtb, tidx, 1, out + OFF_ETGT, Bn * NTPc);                       // etgt
  gather_kernel<<<CDIV((size_t)Bn * HID, 256), 256, 0, stream>>>(
      tgtb, tidx, NTPc, out + OFF_VTGT, Bn);                           // vis_tgt
  gather_kernel<<<CDIV((size_t)Bn * HID, 256), 256, 0, stream>>>(
      emb, tidx, NTPc, out + OFF_VINIT, Bn);                           // vis_init_tgt
  masked_mean_kernel<<<Bn, 128, 0, stream>>>(tgtb, Bv(IDX_MASK), out + OFF_VGRPH);
  copy_kernel<<<CDIV(Bn * HID, 256), 256, 0, stream>>>(ectx, out + OFF_VCTX1, Bn * HID);
  copy_kernel<<<CDIV(Bn * HID, 256), 256, 0, stream>>>(ectx, out + OFF_VCTX2, Bn * HID);

  // predictor: cond = ectx(broadcast) + relu(pe[tgt_idx] @ Wprw + b)
  gemm(stream, F(IDX_PE), 16, tidx, nullptr, nullptr, 0, Wprw16, 16, 128,
       F(PRWB), nullptr, nullptr, nullptr, nullptr, tpe, Bn * NTPc, 1);
  bcast_add_kernel<<<CDIV((size_t)Bn * NTPc * HID, 256), 256, 0, stream>>>(ectx, tpe, cond);
  gemm(stream, cond, 128, nullptr, nullptr, nullptr, 0, Wl1, 128, 128, F(PL1B),
       nullptr, nullptr, nullptr, nullptr, predh, Bn * NTPc, 0);
  ln_kernel<<<CDIV(Bn * NTPc, 256), 256, 0, stream>>>(predh, predh, F(PLNG), F(PLNB), Bn * NTPc, 1);
  gemm(stream, predh, 128, nullptr, nullptr, nullptr, 0, Wl2, 128, 128, F(PL2B),
       nullptr, nullptr, nullptr, nullptr, out + OFF_PRED, Bn * NTPc, 0);
}
